// TopologicalLoss_14620068675661
// MI455X (gfx1250) — compile-verified, matrix operand order checked by reading, not run on hardware
//
#include <hip/hip_runtime.h>
#include <hip/hip_bf16.h>

// ---------------------------------------------------------------------------
// TopologicalLoss on MI455X (gfx1250):
//   S = Zg (2048x512) . Zk^T (512x32768)  -> bf16 WMMA, fp32 accumulate
//   d2 = |x|^2 + |y|^2 - 2S ; d = sqrt(max(d2,0)); w = exp(-d2)
//   soft = sum(w*d)/(sum(w)+eps) per row; loss = -mean(min(soft, 3))
//
// GEMM structure (compute-bound: ~1000 FLOP/byte vs HBM; must feed WMMA):
//   - 128x128 macro-tile per workgroup, K staged in 32-wide panels
//   - A/B panels (8 KB each) double-buffered in LDS via
//     GLOBAL_LOAD_ASYNC_TO_LDS_B128 (ASYNCcnt + s_wait_asynccnt)
//   - 8 waves, each owns a 64x32 register tile (4x2 C fragments):
//     per K-step: 6 LDS fragment reads -> 8 v_wmma_f32_16x16x32_bf16
//   - fused RBF epilogue, register accumulation across 4 N-macro-tiles,
//     16-lane butterfly reduce, one fp32 atomicAdd pair per row
// ---------------------------------------------------------------------------

typedef __attribute__((ext_vector_type(16))) __bf16 v16bf;
typedef __attribute__((ext_vector_type(8)))  float  v8f;
typedef __attribute__((ext_vector_type(4)))  int    v4i;

// Pointer-to-int4 in global (AS1) / LDS (AS3) address spaces, matching the
// probed signature of __builtin_amdgcn_global_load_async_to_lds_b128.
typedef __attribute__((address_space(1))) v4i* gv4i_p;
typedef __attribute__((address_space(3))) v4i* lv4i_p;

#define NGEN   2048      // rows of z_generated (M)
#define NKNOWN 32768     // rows of z_known     (N)
#define DIM    512       // feature dim         (K)
#define MTILES (NGEN / 16)
#define NTILES (NKNOWN / 16)
#define KTILES (DIM / 32)
#define NMACRO (NKNOWN / 128)     // 256 N macro-tiles
#define NCHUNK 64                 // grid.y; each block walks NMACRO/NCHUNK = 4

#define MAX_RADIUS 3.0f
#define EPS        1e-8f

// ----------------------- CDNA5 async-copy primitives ------------------------

static __device__ __forceinline__ void async_copy_b128(const unsigned short* g,
                                                       unsigned short* l) {
#if __has_builtin(__builtin_amdgcn_global_load_async_to_lds_b128)
    __builtin_amdgcn_global_load_async_to_lds_b128(
        (gv4i_p)(unsigned long long)(const void*)g,
        (lv4i_p)(unsigned)(unsigned long long)(void*)l,
        0, 0);
#else
    // VDST = LDS byte address, VADDR = 64-bit global address (ISA 15.18.3 op 98)
    asm volatile("global_load_async_to_lds_b128 %0, %1, off"
                 :: "v"((unsigned)(unsigned long long)(void*)l),
                    "v"((unsigned long long)(const void*)g)
                 : "memory");
#endif
}

#if __has_builtin(__builtin_amdgcn_s_wait_asynccnt)
#define WAIT_ASYNC(n) __builtin_amdgcn_s_wait_asynccnt(n)
#else
#define WAIT_ASYNC(n) asm volatile("s_wait_asynccnt %0" :: "n"(n) : "memory")
#endif

// -------------------------- helpers ----------------------------------------

static __device__ __forceinline__ unsigned short f32_to_bf16_rne(float f) {
    unsigned int u = __float_as_uint(f);
    unsigned int r = 0x7FFFu + ((u >> 16) & 1u);   // round-to-nearest-even
    u += r;
    return (unsigned short)(u >> 16);
}

// -------------------------- pack kernel ------------------------------------
// fp32 -> bf16, pre-swizzled into the CDNA5 16-bit A-fragment lane layout
// (ISA 7.12.2, 16-bit A 16x32):
//   lane l (0..31): row = l%16; K-base = (l>=16 ? 8 : 0)
//   elem e (0..15): v = e>>1; koff = (v<4) ? 2v : 16+2(v-4); k = base+koff+(e&1)
// Output: frag[((rowTile*KTILES + kTile)*32 + lane)*16 + e]  (32B per lane)
__global__ void topo_pack_bf16_frag(const float* __restrict__ src,
                                    unsigned short* __restrict__ dst,
                                    int rowTiles) {
    int t = blockIdx.x * blockDim.x + threadIdx.x;
    int rt = t >> 13;                        // 16*32*16 = 8192 elems per row-tile
    if (rt >= rowTiles) return;
    int e    = t & 15;
    int lane = (t >> 4) & 31;
    int kt   = (t >> 9) & 15;

    int row  = rt * 16 + (lane & 15);
    int base = (lane >> 4) * 8;
    int v    = e >> 1;
    int koff = (v < 4) ? (2 * v) : (16 + 2 * (v - 4));
    int k    = kt * 32 + base + koff + (e & 1);

    dst[t] = f32_to_bf16_rne(src[row * DIM + k]);
}

// -------------------------- row-norm kernel ---------------------------------
__global__ void topo_row_norm2(const float* __restrict__ x,
                               float* __restrict__ out, int rows) {
    int wave = (blockIdx.x * blockDim.x + threadIdx.x) >> 5;
    int lane = threadIdx.x & 31;
    if (wave >= rows) return;
    const float4* p = (const float4*)(x + (size_t)wave * DIM);
    float acc = 0.0f;
    #pragma unroll
    for (int i = lane; i < DIM / 4; i += 32) {
        float4 v = p[i];
        acc += v.x * v.x + v.y * v.y + v.z * v.z + v.w * v.w;
    }
    #pragma unroll
    for (int m = 16; m >= 1; m >>= 1) acc += __shfl_xor(acc, m, 32);
    if (lane == 0) out[wave] = acc;
}

// -------------------------- init kernel -------------------------------------
__global__ void topo_init_sums(float* __restrict__ sum_w,
                               float* __restrict__ sum_wd) {
    int i = blockIdx.x * blockDim.x + threadIdx.x;
    if (i < NGEN) { sum_w[i] = 0.0f; sum_wd[i] = 0.0f; }
}

// -------------------------- fused GEMM + epilogue ----------------------------
// grid = (MTILES/8, NCHUNK) = (16, 64); block = 256 (8 wave32's).
__global__ void __launch_bounds__(256)
topo_gemm_rbf(const unsigned short* __restrict__ Af,
              const unsigned short* __restrict__ Bf,
              const float* __restrict__ x2,
              const float* __restrict__ y2,
              float* __restrict__ sum_w,
              float* __restrict__ sum_wd) {
    // Double-buffered bf16 panels: [buf][8 subtiles][32 lanes][16 elems]
    __shared__ __align__(32) unsigned short ldsA[2][8 * 512];
    __shared__ __align__(32) unsigned short ldsB[2][8 * 512];

    const int lane   = threadIdx.x & 31;
    const int waveId = threadIdx.x >> 5;
    const int sub    = waveId;            // staging subtile for this wave's lanes
    const int wm     = waveId >> 2;       // 0..1 : 64-row group
    const int wn     = waveId & 3;        // 0..3 : 32-col group
    const int half   = lane >> 4;         // C layout: lanes 16..31 hold M = vg+8
    const int ncol   = lane & 15;         // C layout: N = lane % 16

    const int rowBase = blockIdx.x * 128 + wm * 64;

    // Global staging sources (per-thread, 32 B each): A subtile 'sub' of this
    // macro row; B subtile 'sub' of the current N macro-tile.
    const unsigned short* aSrc =
        Af + ((size_t)(blockIdx.x * 8 + sub) * KTILES) * 512 + lane * 16;

    // x2 for the 32 rows this lane touches (invariant over N)
    float x2v[4][8];
    #pragma unroll
    for (int i = 0; i < 4; ++i)
        #pragma unroll
        for (int vg = 0; vg < 8; ++vg)
            x2v[i][vg] = x2[rowBase + i * 16 + vg + 8 * half];

    float aw[4][8], awd[4][8];
    #pragma unroll
    for (int i = 0; i < 4; ++i)
        #pragma unroll
        for (int vg = 0; vg < 8; ++vg) { aw[i][vg] = 0.0f; awd[i][vg] = 0.0f; }

    for (int nmi = 0; nmi < NMACRO / NCHUNK; ++nmi) {
        const int nm = blockIdx.y + nmi * NCHUNK;          // 0..255
        const unsigned short* bSrc =
            Bf + ((size_t)(nm * 8 + sub) * KTILES) * 512 + lane * 16;

        v8f c[4][2];
        #pragma unroll
        for (int i = 0; i < 4; ++i)
            #pragma unroll
            for (int j = 0; j < 2; ++j) { v8f z = {}; c[i][j] = z; }

        // prologue: stage kt=0 into buffer 0
        {
            unsigned short* la = &ldsA[0][sub * 512 + lane * 16];
            unsigned short* lb = &ldsB[0][sub * 512 + lane * 16];
            async_copy_b128(aSrc,     la);
            async_copy_b128(aSrc + 8, la + 8);
            async_copy_b128(bSrc,     lb);
            async_copy_b128(bSrc + 8, lb + 8);
        }

        for (int kt = 0; kt < KTILES; ++kt) {
            const int buf = kt & 1;
            if (kt + 1 < KTILES) {
                const int nb = (kt + 1) & 1;
                const unsigned short* ga = aSrc + (size_t)(kt + 1) * 512;
                const unsigned short* gb = bSrc + (size_t)(kt + 1) * 512;
                unsigned short* la = &ldsA[nb][sub * 512 + lane * 16];
                unsigned short* lb = &ldsB[nb][sub * 512 + lane * 16];
                async_copy_b128(ga,     la);
                async_copy_b128(ga + 8, la + 8);
                async_copy_b128(gb,     lb);
                async_copy_b128(gb + 8, lb + 8);
                WAIT_ASYNC(4);       // stage kt retired; 4 just-issued may fly
            } else {
                WAIT_ASYNC(0);
            }
            __syncthreads();         // stage kt visible to all waves

            v16bf afrag[4], bfrag[2];
            #pragma unroll
            for (int i = 0; i < 4; ++i)
                afrag[i] = *(const v16bf*)&ldsA[buf][(wm * 4 + i) * 512 + lane * 16];
            #pragma unroll
            for (int j = 0; j < 2; ++j)
                bfrag[j] = *(const v16bf*)&ldsB[buf][(wn * 2 + j) * 512 + lane * 16];

            #pragma unroll
            for (int i = 0; i < 4; ++i)
                #pragma unroll
                for (int j = 0; j < 2; ++j)
                    c[i][j] = __builtin_amdgcn_wmma_f32_16x16x32_bf16(
                        false, afrag[i], false, bfrag[j],
                        (short)0, c[i][j], false, false);

            __syncthreads();         // all reads of buf done before overwrite
        }

        // fused RBF epilogue for this wave's 64x32 tile
        const int nBase = nm * 128 + wn * 32;
        #pragma unroll
        for (int j = 0; j < 2; ++j) {
            float y2v = y2[nBase + j * 16 + ncol];
            #pragma unroll
            for (int i = 0; i < 4; ++i) {
                #pragma unroll
                for (int vg = 0; vg < 8; ++vg) {
                    float d2 = x2v[i][vg] + y2v - 2.0f * c[i][j][vg];
                    d2 = fmaxf(d2, 0.0f);
                    float d = __builtin_sqrtf(d2);
                    float w = __expf(-d2);
                    aw[i][vg]  += w;
                    awd[i][vg] += w * d;
                }
            }
        }
    }

    // 16-lane butterfly reduce (lanes sharing an M row), then atomic combine.
    #pragma unroll
    for (int i = 0; i < 4; ++i) {
        #pragma unroll
        for (int vg = 0; vg < 8; ++vg) {
            float a = aw[i][vg], b = awd[i][vg];
            #pragma unroll
            for (int m = 8; m >= 1; m >>= 1) {
                a += __shfl_xor(a, m, 32);
                b += __shfl_xor(b, m, 32);
            }
            if (ncol == 0) {
                int row = rowBase + i * 16 + vg + 8 * half;
                atomicAdd(&sum_w[row],  a);
                atomicAdd(&sum_wd[row], b);
            }
        }
    }
}

// -------------------------- finalize kernel ----------------------------------
__global__ void topo_finalize(const float* __restrict__ sum_w,
                              const float* __restrict__ sum_wd,
                              float* __restrict__ out) {
    __shared__ float red[256];
    float acc = 0.0f;
    for (int i = threadIdx.x; i < NGEN; i += 256) {
        float soft = sum_wd[i] / (sum_w[i] + EPS);
        acc += fminf(soft, MAX_RADIUS);
    }
    red[threadIdx.x] = acc;
    __syncthreads();
    for (int off = 128; off > 0; off >>= 1) {
        if (threadIdx.x < off) red[threadIdx.x] += red[threadIdx.x + off];
        __syncthreads();
    }
    if (threadIdx.x == 0) out[0] = -red[0] / (float)NGEN;
}

// -------------------------- launch ------------------------------------------
extern "C" void kernel_launch(void* const* d_in, const int* in_sizes, int n_in,
                              void* d_out, int out_size, void* d_ws, size_t ws_size,
                              hipStream_t stream) {
    const float* zg = (const float*)d_in[0];   // [2048, 512] fp32
    const float* zk = (const float*)d_in[1];   // [32768, 512] fp32
    float* out = (float*)d_out;

    // Workspace layout (~34.2 MB total)
    char* ws = (char*)d_ws;
    unsigned short* Af = (unsigned short*)ws;                        //  2 MB
    unsigned short* Bf = (unsigned short*)(ws + (size_t)(2u << 20)); // 32 MB
    char* tail = ws + (size_t)(34u << 20);
    float* x2     = (float*)(tail);                      // 8 KB
    float* y2     = (float*)(tail + 8192);               // 128 KB
    float* sum_w  = (float*)(tail + 8192 + 131072);      // 8 KB
    float* sum_wd = (float*)(tail + 8192 + 131072 + 8192);

    // 1) zero accumulators (deterministic per call; graph-capture safe)
    topo_init_sums<<<NGEN / 256, 256, 0, stream>>>(sum_w, sum_wd);

    // 2) bf16 pack + swizzle into WMMA fragment order
    topo_pack_bf16_frag<<<(NGEN * DIM) / 256, 256, 0, stream>>>(zg, Af, MTILES);
    topo_pack_bf16_frag<<<(NKNOWN * DIM) / 256, 256, 0, stream>>>(zk, Bf, NTILES);

    // 3) fp32 row norms
    topo_row_norm2<<<(NGEN * 32) / 256, 256, 0, stream>>>(zg, x2, NGEN);
    topo_row_norm2<<<(NKNOWN * 32) / 256, 256, 0, stream>>>(zk, y2, NKNOWN);

    // 4) fused WMMA GEMM + RBF epilogue (LDS double-buffered async staging)
    dim3 grid(MTILES / 8, NCHUNK);
    topo_gemm_rbf<<<grid, 256, 0, stream>>>(Af, Bf, x2, y2, sum_w, sum_wd);

    // 5) scalar loss
    topo_finalize<<<1, 256, 0, stream>>>(sum_w, sum_wd, out);
}
